// SelfAttentionWithRoPE_76184129896564
// MI455X (gfx1250) — matrix operand order repaired and checked by reading, not compiled, not run on hardware
//
#include <hip/hip_runtime.h>

typedef __attribute__((ext_vector_type(16))) _Float16 v16h;
typedef __attribute__((ext_vector_type(8)))  _Float16 v8h;
typedef __attribute__((ext_vector_type(8)))  float    v8f;

// Problem constants (fixed by reference)
#define BB 2
#define SS 2048
#define DD 2048
#define HH 16
#define DH 128
#define ND3 (3*DD)
#define MTOT (BB*SS)          // 4096
static_assert(DD == HH*DH, "");
static_assert((DD & 63) == 0, "K loop unrolled by 64");

// ---------------- WMMA helpers (CDNA5 16x16x32 f16 -> f32) ----------------

__device__ __forceinline__ v8f wmma_f16(v16h a, v16h b, v8f c) {
  // (neg_a, A, neg_b, B, c_mod, C, reuse_a, reuse_b)
  return __builtin_amdgcn_wmma_f32_16x16x32_f16(false, a, false, b, (short)0, c,
                                                false, false);
}

// A-matrix fragment (16x32, f16 row-major, row stride ld).
// Lane l: M = l&15 ; 16 contiguous K halves starting at kk + half*16.
__device__ __forceinline__ v16h load_a_frag(const _Float16* base, int ld, int kk) {
  const int lane = threadIdx.x & 31;
  const int row  = lane & 15;
  const int half = lane >> 4;
  const _Float16* p = base + row * ld + kk + half * 16;
  v8h lo = *(const v8h*)p;
  v8h hi = *(const v8h*)(p + 8);
  return __builtin_shufflevector(lo, hi, 0,1,2,3,4,5,6,7,8,9,10,11,12,13,14,15);
}

// B-matrix fragment (32x16): operand stored N-major (W[n][k], row stride ld).
// Lane l: N = l&15 ; 16 contiguous K halves starting at kk + (l>>4)*16.
__device__ __forceinline__ v16h load_b_frag(const _Float16* base, int ld, int kk) {
  const int lane = threadIdx.x & 31;
  const int n    = lane & 15;
  const int koff = (lane >> 4) * 16;
  return *(const v16h*)(base + n * ld + kk + koff);
}

// ---------------- Kernel 0: fp32 -> f16 conversion ----------------

__global__ void cvt_f32_to_f16(const float* __restrict__ in,
                               _Float16* __restrict__ out, int n) {
  int i = blockIdx.x * blockDim.x + threadIdx.x;
  int stride = gridDim.x * blockDim.x;
  for (; i < n; i += stride) out[i] = (_Float16)in[i];
}

// ---------------- Pipelined 64x32 wave-tile GEMM core ----------------
// C[m][n] = sum_k A[m][k] * B[n][k]. Ping-pong double buffering with
// statically alternating register sets (no rotation copies).

__device__ __forceinline__ void gemm_wave_64x32(
    const _Float16* __restrict__ Abase,   // rows m0..m0+63, stride DD
    const _Float16* __restrict__ Bbase,   // rows n0..n0+31, stride DD
    int col0, v8f acc[4][2]) {
  v16h a0[4], b0[2], a1[4], b1[2];
#pragma unroll
  for (int mt = 0; mt < 4; ++mt) a0[mt] = load_a_frag(Abase + mt * 16 * DD, DD, 0);
#pragma unroll
  for (int nt = 0; nt < 2; ++nt) b0[nt] = load_b_frag(Bbase + nt * 16 * DD, DD, 0);

  for (int kk = 0; kk < DD; kk += 64) {
    // load stage-1 fragments (kk+32 always < DD)
#pragma unroll
    for (int mt = 0; mt < 4; ++mt)
      a1[mt] = load_a_frag(Abase + mt * 16 * DD, DD, kk + 32);
#pragma unroll
    for (int nt = 0; nt < 2; ++nt)
      b1[nt] = load_b_frag(Bbase + nt * 16 * DD, DD, kk + 32);
    if (kk + 192 < DD) {
      __builtin_prefetch(Abase + col0 * DD + kk + 192, 0, 2);
      __builtin_prefetch(Bbase + col0 * DD + kk + 192, 0, 2);
    }
    // compute on stage-0
#pragma unroll
    for (int mt = 0; mt < 4; ++mt)
#pragma unroll
      for (int nt = 0; nt < 2; ++nt)
        acc[mt][nt] = wmma_f16(a0[mt], b0[nt], acc[mt][nt]);
    // load stage-0 fragments for next trip
    if (kk + 64 < DD) {
#pragma unroll
      for (int mt = 0; mt < 4; ++mt)
        a0[mt] = load_a_frag(Abase + mt * 16 * DD, DD, kk + 64);
#pragma unroll
      for (int nt = 0; nt < 2; ++nt)
        b0[nt] = load_b_frag(Bbase + nt * 16 * DD, DD, kk + 64);
    }
    // compute on stage-1
#pragma unroll
    for (int mt = 0; mt < 4; ++mt)
#pragma unroll
      for (int nt = 0; nt < 2; ++nt)
        acc[mt][nt] = wmma_f16(a1[mt], b1[nt], acc[mt][nt]);
  }
}

// ---------------- Kernel 1: QKV GEMM + RoPE + layout ----------------
// m in [0,4096), n in [0,6144); block = 8 waves, tile 64(M) x 256(N).

__global__ __launch_bounds__(256) void qkv_rope_kernel(
    const _Float16* __restrict__ Xh, const _Float16* __restrict__ Wh,
    _Float16* __restrict__ Qh, _Float16* __restrict__ Kh,
    _Float16* __restrict__ Vth) {
  const int wave = threadIdx.x >> 5;
  const int lane = threadIdx.x & 31;
  const int half = lane >> 4;
  const int col0 = lane & 15;
  const int m0 = blockIdx.x * 64;
  const int n0 = blockIdx.y * 256 + wave * 32;

  v8f acc[4][2] = {};
  gemm_wave_64x32(Xh + (size_t)m0 * DD, Wh + (size_t)n0 * DD, col0, acc);

  // Epilogue: decode n -> (which,h,d); RoPE on q,k; scatter to attention layouts.
#pragma unroll
  for (int mt = 0; mt < 4; ++mt)
#pragma unroll
    for (int nt = 0; nt < 2; ++nt)
#pragma unroll
      for (int v = 0; v < 8; ++v) {
        float val = acc[mt][nt][v];
        float pv  = __shfl_xor(val, 1, 32);  // RoPE pair partner (d^1, same row)
        const int m = m0 + mt * 16 + v + half * 8;
        const int n = n0 + nt * 16 + col0;
        const int which = n >> 11;          // 0=q 1=k 2=v
        const int h = (n >> 7) & 15;
        const int d = n & 127;
        const int b_ = m >> 11;
        const int s  = m & 2047;
        if (which <= 1) {
          const int i = d >> 1;
          // inv_freq = 10000^(-2i/128) = exp(-i * ln(10000)/64)
          float inv = __expf(-(float)i * 0.14391156834337935f);
          float ang = (float)s * inv;
          float sn, cs;
          __sincosf(ang, &sn, &cs);
          val = (d & 1) ? fmaf(val, cs, pv * sn) : fmaf(val, cs, -pv * sn);
        }
        const _Float16 hv = (_Float16)val;
        const int bh = b_ * HH + h;
        if (which == 0)      Qh[(bh * SS + s) * DH + d] = hv;
        else if (which == 1) Kh[(bh * SS + s) * DH + d] = hv;
        else                 Vth[(bh * DH + d) * SS + s] = hv;  // V transposed
      }
}

// ---------------- Kernel 2: causal flash attention ----------------
// One wave (32 threads) per (b,h, 16-row query tile). 4096 blocks.

__global__ __launch_bounds__(32) void attn_kernel(
    const _Float16* __restrict__ Qh, const _Float16* __restrict__ Kh,
    const _Float16* __restrict__ Vth, _Float16* __restrict__ AOh) {
  __shared__ __align__(32) _Float16 plds[16 * 32];  // P tile transpose buffer

  const int lane = threadIdx.x & 31;
  const int half = lane >> 4;
  const int col0 = lane & 15;
  const int gw = blockIdx.x;
  const int bh = gw >> 7;             // (b*16+h)
  const int q0 = (gw & 127) * 16;
  const int b_ = bh >> 4;
  const int h  = bh & 15;

  const _Float16* Qb = Qh + (size_t)bh * SS * DH;
  const _Float16* Kb = Kh + (size_t)bh * SS * DH;
  const _Float16* Vb = Vth + (size_t)bh * DH * SS;

  v16h qf[4];
#pragma unroll
  for (int kt = 0; kt < 4; ++kt) qf[kt] = load_a_frag(Qb + q0 * DH, DH, kt * 32);

  v8f o[8] = {};
  float mrow[8], lrow[8];
#pragma unroll
  for (int v = 0; v < 8; ++v) { mrow[v] = -1e30f; lrow[v] = 0.f; }

  const int nblocks = (q0 + 15) / 32 + 1;  // causal: keys <= q0+15
  for (int j = 0; j < nblocks; ++j) {
    // S = Q @ K^T for 16 q-rows x 32 keys
    v8f sf[2] = {};
#pragma unroll
    for (int kt = 0; kt < 4; ++kt) {
      v16h kb0 = load_b_frag(Kb + (j * 32) * DH, DH, kt * 32);
      v16h kb1 = load_b_frag(Kb + (j * 32 + 16) * DH, DH, kt * 32);
      sf[0] = wmma_f16(qf[kt], kb0, sf[0]);
      sf[1] = wmma_f16(qf[kt], kb1, sf[1]);
    }
    // Hoist V-fragment loads: independent of softmax math below, so these
    // global loads overlap the VALU-heavy online-softmax section.
    v16h vf[8];
#pragma unroll
    for (int dt = 0; dt < 8; ++dt)
      vf[dt] = load_b_frag(Vb + dt * 16 * SS, SS, j * 32);

    // scale + causal mask
#pragma unroll
    for (int nt = 0; nt < 2; ++nt)
#pragma unroll
      for (int v = 0; v < 8; ++v) {
        const int kcol = j * 32 + nt * 16 + col0;
        const int qrow = q0 + v + half * 8;
        float sv = sf[nt][v] * 0.08838834764831845f;  // 1/sqrt(128)
        sf[nt][v] = (kcol > qrow) ? -1e30f : sv;
      }
    // block row-max (reduce across the 16 lanes sharing each row)
    float bm[8];
#pragma unroll
    for (int v = 0; v < 8; ++v) bm[v] = fmaxf(sf[0][v], sf[1][v]);
#pragma unroll
    for (int off = 1; off < 16; off <<= 1)
#pragma unroll
      for (int v = 0; v < 8; ++v) bm[v] = fmaxf(bm[v], __shfl_xor(bm[v], off, 32));
    // online-softmax rescale
    float psum[8];
#pragma unroll
    for (int v = 0; v < 8; ++v) {
      float mnew  = fmaxf(mrow[v], bm[v]);
      float alpha = __expf(mrow[v] - mnew);
      mrow[v] = mnew;
      lrow[v] *= alpha;
#pragma unroll
      for (int dt = 0; dt < 8; ++dt) o[dt][v] *= alpha;
      psum[v] = 0.f;
    }
    // P = exp(S - m), accumulate row sums, stage to LDS (C-layout -> A-layout)
#pragma unroll
    for (int nt = 0; nt < 2; ++nt)
#pragma unroll
      for (int v = 0; v < 8; ++v) {
        float p = __expf(sf[nt][v] - mrow[v]);
        psum[v] += p;
        plds[(v + half * 8) * 32 + nt * 16 + col0] = (_Float16)p;
      }
#pragma unroll
    for (int off = 1; off < 16; off <<= 1)
#pragma unroll
      for (int v = 0; v < 8; ++v) psum[v] += __shfl_xor(psum[v], off, 32);
#pragma unroll
    for (int v = 0; v < 8; ++v) lrow[v] += psum[v];

    __syncthreads();                         // single-wave WG: waits DS, barrier=NOP
    v16h pf = load_a_frag(plds, 32, 0);      // P as A-fragment (16x32)
    __syncthreads();

    // O += P @ V  (B operand from V-transposed: contiguous in s)
#pragma unroll
    for (int dt = 0; dt < 8; ++dt) o[dt] = wmma_f16(pf, vf[dt], o[dt]);
  }

  // normalize and store attention output as (b, s, h*128+d) f16
  float invl[8];
#pragma unroll
  for (int v = 0; v < 8; ++v) invl[v] = 1.0f / lrow[v];
#pragma unroll
  for (int dt = 0; dt < 8; ++dt)
#pragma unroll
    for (int v = 0; v < 8; ++v) {
      const int s = q0 + v + half * 8;
      const int d = dt * 16 + col0;
      AOh[((size_t)(b_ * SS + s)) * DD + h * DH + d] = (_Float16)(o[dt][v] * invl[v]);
    }
}

// ---------------- Kernel 3: output projection ----------------
// out[m][n] = sum_k AOh[m][k] * Wout[n][k]; fp32 output.

__global__ __launch_bounds__(256) void outproj_kernel(
    const _Float16* __restrict__ AOh, const _Float16* __restrict__ Wh,
    float* __restrict__ out) {
  const int wave = threadIdx.x >> 5;
  const int lane = threadIdx.x & 31;
  const int half = lane >> 4;
  const int col0 = lane & 15;
  const int m0 = blockIdx.x * 64;
  const int n0 = blockIdx.y * 256 + wave * 32;

  v8f acc[4][2] = {};
  gemm_wave_64x32(AOh + (size_t)m0 * DD, Wh + (size_t)n0 * DD, col0, acc);

#pragma unroll
  for (int mt = 0; mt < 4; ++mt)
#pragma unroll
    for (int nt = 0; nt < 2; ++nt)
#pragma unroll
      for (int v = 0; v < 8; ++v) {
        const int m = m0 + mt * 16 + v + half * 8;
        const int n = n0 + nt * 16 + col0;
        out[(size_t)m * DD + n] = acc[mt][nt][v];
      }
}

// ---------------- launcher ----------------

extern "C" void kernel_launch(void* const* d_in, const int* in_sizes, int n_in,
                              void* d_out, int out_size, void* d_ws, size_t ws_size,
                              hipStream_t stream) {
  const float* x     = (const float*)d_in[0];   // (2,2048,2048)
  const float* w_qkv = (const float*)d_in[1];   // (6144,2048)
  const float* w_out = (const float*)d_in[2];   // (2048,2048)
  float* out = (float*)d_out;

  // Workspace carve (f16 staging, 112 MB total, L2-resident)
  char* w = (char*)d_ws;
  _Float16* Xh    = (_Float16*)w;                          // 16 MB
  _Float16* Wqkvh = (_Float16*)(w + (((size_t)16) << 20)); // 24 MB
  _Float16* Wouth = (_Float16*)(w + (((size_t)40) << 20)); //  8 MB
  _Float16* Qh    = (_Float16*)(w + (((size_t)48) << 20)); // 16 MB
  _Float16* Kh    = (_Float16*)(w + (((size_t)64) << 20)); // 16 MB
  _Float16* Vth   = (_Float16*)(w + (((size_t)80) << 20)); // 16 MB
  _Float16* AOh   = (_Float16*)(w + (((size_t)96) << 20)); // 16 MB

  cvt_f32_to_f16<<<2048, 256, 0, stream>>>(x,     Xh,    BB * SS * DD);
  cvt_f32_to_f16<<<2048, 256, 0, stream>>>(w_qkv, Wqkvh, ND3 * DD);
  cvt_f32_to_f16<<<2048, 256, 0, stream>>>(w_out, Wouth, DD * DD);

  qkv_rope_kernel<<<dim3(MTOT / 64, ND3 / 256), 256, 0, stream>>>(
      Xh, Wqkvh, Qh, Kh, Vth);

  attn_kernel<<<BB * HH * (SS / 16), 32, 0, stream>>>(Qh, Kh, Vth, AOh);

  outproj_kernel<<<dim3(MTOT / 64, DD / 256), 256, 0, stream>>>(AOh, Wouth, out);
}
